// MotifAttentionLayer_3582002724977
// MI455X (gfx1250) — matrix operand order repaired and checked by reading, not compiled, not run on hardware
//
#include <hip/hip_runtime.h>
#include <hip/hip_bf16.h>

// MotifAttentionLayer for MI455X (gfx1250):
//   bf16 WMMA GEMMs + conv-as-WMMA + TDM (tensor_load_to_lds) tile staging.
// B=8 S=4096 H=768 NH=12 HD=64 K=7

#define BB     8
#define SS     4096
#define HH     768
#define NHEADS 12
#define HDIM   64
#define KCONV  7
#define MROWS  (BB * SS)   // 32768
#define QKVN   (3 * HH)    // 2304
#define LN_EPS 1e-12f

typedef __attribute__((ext_vector_type(16))) __bf16 v16bf;
typedef __attribute__((ext_vector_type(8)))  float  v8f;
typedef unsigned int       u32;
typedef unsigned short     u16;
typedef unsigned long long u64;

#ifndef __has_builtin
#define __has_builtin(x) 0
#endif
#if __has_builtin(__builtin_amdgcn_tensor_load_to_lds)
#define USE_TDM 1
#else
#define USE_TDM 0
#endif

union FragU { uint4 q[2]; v16bf v; };

__device__ __forceinline__ u16 f2bf(float f) {
    union { float f; u32 u; } c; c.f = f;
    u32 r = c.u + 0x7FFFu + ((c.u >> 16) & 1u);   // round-to-nearest-even
    return (u16)(r >> 16);
}
__device__ __forceinline__ float bf2f(u16 h) {
    union { u32 u; float f; } c; c.u = ((u32)h) << 16;
    return c.f;
}

#if USE_TDM
typedef __attribute__((ext_vector_type(4))) u32 tdm_v4u;
typedef __attribute__((ext_vector_type(8))) int tdm_v8i;
typedef __attribute__((ext_vector_type(4))) int tdm_v4i;

// 2D tile DMA: global (row-major, strideElems between rows, 2-byte elems)
// -> LDS contiguous [tileH][tileW]. D# per CDNA5 ISA ch.8 (group0/group1).
__device__ __forceinline__ void tdm_load_2d(u32 lds_byte, const void* gsrc,
                                            u32 tileW, u32 tileH, u32 strideElems) {
    u64 ga = (u64)(size_t)gsrc;
    tdm_v4u g0;
    g0.x = 1u;                                              // count=1 valid, user mode
    g0.y = lds_byte;                                        // lds_addr (bytes)
    g0.z = (u32)ga;                                         // global_addr[31:0]
    g0.w = ((u32)(ga >> 32) & 0x01FFFFFFu) | (2u << 30);    // global_addr[56:32], type=2
    const u32 dim0 = 0x40000000u, dim1 = 0x40000000u;       // huge tensor dims: tiles never clip
    tdm_v8i g1;
    g1[0] = (int)(1u << 16);                                // workgroup_mask=0, data_size=1 (2B)
    g1[1] = (int)((dim0 & 0xFFFFu) << 16);                  // tensor_dim0[15:0] @ bits 63:48
    g1[2] = (int)((dim0 >> 16) | ((dim1 & 0xFFFFu) << 16)); // tensor_dim0[31:16], tensor_dim1[15:0]
    g1[3] = (int)(((dim1 >> 16) & 0xFFFFu) | (tileW << 16));// tensor_dim1[31:16], tile_dim0
    g1[4] = (int)(tileH & 0xFFFFu);                         // tile_dim1 (tile_dim2=0 -> 2D)
    g1[5] = (int)strideElems;                               // tensor_dim0_stride[31:0]
    g1[6] = 0;                                              // stride0[47:32]=0, stride1[15:0]=0
    g1[7] = 0;
    tdm_v4i gz = {0, 0, 0, 0};
#if __clang_major__ >= 23
    tdm_v8i gz8 = {0, 0, 0, 0, 0, 0, 0, 0};
    __builtin_amdgcn_tensor_load_to_lds(g0, g1, gz, gz, gz8, 0);
#else
    __builtin_amdgcn_tensor_load_to_lds(g0, g1, gz, gz, 0);
#endif
}
#endif  // USE_TDM

// ---------------------------------------------------------------- pack kernels
__global__ void pack_x_kernel(const float* __restrict__ x, u16* __restrict__ xb, int n) {
    int i = blockIdx.x * blockDim.x + threadIdx.x;
    if (i < n) xb[i] = f2bf(x[i]);
}

// WqkvT[n][k] = W{q,k,v}[k][n%768]; bqkv[n] = b{q,k,v}[n%768]
__global__ void pack_wqkv_kernel(const float* __restrict__ Wq, const float* __restrict__ Wk,
                                 const float* __restrict__ Wv, const float* __restrict__ bq,
                                 const float* __restrict__ bk, const float* __restrict__ bv,
                                 u16* __restrict__ WT, float* __restrict__ bqkv) {
    int i = blockIdx.x * blockDim.x + threadIdx.x;
    if (i >= QKVN * HH) return;
    int n = i / HH, k = i - n * HH;
    const float* W = (n < HH) ? Wq : (n < 2 * HH ? Wk : Wv);
    int nn = (n < HH) ? n : (n < 2 * HH ? n - HH : n - 2 * HH);
    WT[i] = f2bf(W[k * HH + nn]);
    if (k == 0) {
        const float* b = (n < HH) ? bq : (n < 2 * HH ? bk : bv);
        bqkv[n] = b[nn];
    }
}

__global__ void pack_wo_kernel(const float* __restrict__ Wo, u16* __restrict__ WoT) {
    int i = blockIdx.x * blockDim.x + threadIdx.x;
    if (i >= HH * HH) return;
    int n = i / HH, k = i - n * HH;
    WoT[i] = f2bf(Wo[k * HH + n]);
}

// cw[t][o][i] = conv_w[o][i][t]  (so B-fragments are contiguous in K=i)
__global__ void pack_conv_kernel(const float* __restrict__ w, u16* __restrict__ cw) {
    int i = blockIdx.x * blockDim.x + threadIdx.x;
    if (i >= HH * HDIM * KCONV) return;
    int o = i / (HDIM * KCONV);
    int r = i - o * (HDIM * KCONV);
    int ic = r / KCONV;
    int t  = r - ic * KCONV;
    cw[((size_t)t * HH + o) * HDIM + ic] = f2bf(w[i]);
}

// ---------------------------------------------------------------- GEMM (bf16 WMMA + TDM staging)
// C[m][n] = sum_k A[m][k] * BT[n][k] + bias[n]  (+ resid, f32 out when RESID)
// Block: 256 thr (8 waves). Tile 128x128, BK=32, double-buffered LDS.
template <bool RESID>
__global__ __launch_bounds__(256) void gemm_wmma_kernel(
    const u16* __restrict__ A, const u16* __restrict__ BT,
    const float* __restrict__ bias, const float* __restrict__ resid,
    u16* __restrict__ outb, float* __restrict__ outf, int Kdim, int Ndim)
{
    __shared__ u16 lA[2][128 * 32];
    __shared__ u16 lB[2][128 * 32];

    const int tid   = threadIdx.x;
    const int lane  = tid & 31;
    const int wid   = tid >> 5;
    const int lo    = lane & 15;
    const int hi    = lane >> 4;
    const int waveM = wid & 3;    // 4 waves over M
    const int waveN = wid >> 2;   // 2 waves over N
    const int blockN = blockIdx.x * 128;
    const int blockM = blockIdx.y * 128;

    const int ldRow = tid >> 1;
    const int ldSeg = (tid & 1) * 16;
    (void)ldRow; (void)ldSeg;

    v8f zero = {0.f, 0.f, 0.f, 0.f, 0.f, 0.f, 0.f, 0.f};
    v8f acc[2][4];
#pragma unroll
    for (int i = 0; i < 2; ++i)
#pragma unroll
        for (int j = 0; j < 4; ++j) acc[i][j] = zero;

    const int steps = Kdim / 32;

#if USE_TDM
    if (tid < 32) {  // wave 0 is the DMA producer
        tdm_load_2d((u32)(size_t)&lA[0][0], A + (size_t)blockM * Kdim, 32, 128, (u32)Kdim);
        tdm_load_2d((u32)(size_t)&lB[0][0], BT + (size_t)blockN * Kdim, 32, 128, (u32)Kdim);
    }
#endif

    for (int i = 0; i < steps; ++i) {
        const int cur = i & 1;
#if USE_TDM
        if (tid < 32) {
            if (i + 1 < steps) {   // kick next tiles, then wait for current pair (in-order TENSORcnt)
                const int kn = (i + 1) * 32;
                tdm_load_2d((u32)(size_t)&lA[cur ^ 1][0], A + (size_t)blockM * Kdim + kn, 32, 128, (u32)Kdim);
                tdm_load_2d((u32)(size_t)&lB[cur ^ 1][0], BT + (size_t)blockN * Kdim + kn, 32, 128, (u32)Kdim);
                __builtin_amdgcn_s_wait_tensorcnt(2);
            } else {
                __builtin_amdgcn_s_wait_tensorcnt(0);
            }
        }
        __syncthreads();           // publish DMA'd tiles to all 8 waves
#else
        const int k0 = i * 32;
        __syncthreads();
        {
            const uint4* sa = (const uint4*)(A + (size_t)(blockM + ldRow) * Kdim + k0 + ldSeg);
            uint4* da = (uint4*)(&lA[cur][0] + ldRow * 32 + ldSeg);
            da[0] = sa[0]; da[1] = sa[1];
            const uint4* sb = (const uint4*)(BT + (size_t)(blockN + ldRow) * Kdim + k0 + ldSeg);
            uint4* db = (uint4*)(&lB[cur][0] + ldRow * 32 + ldSeg);
            db[0] = sb[0]; db[1] = sb[1];
        }
        if (k0 + 32 < Kdim) {
            __builtin_prefetch(A + (size_t)(blockM + ldRow) * Kdim + k0 + 32 + ldSeg, 0, 0);
            __builtin_prefetch(BT + (size_t)(blockN + ldRow) * Kdim + k0 + 32 + ldSeg, 0, 0);
        }
        __syncthreads();
#endif

        FragU a[2];
#pragma unroll
        for (int ii = 0; ii < 2; ++ii) {
            int row = waveM * 32 + ii * 16 + lo;
            const uint4* p = (const uint4*)(&lA[cur][0] + row * 32);
            // A 16x32 bf16 frag, this code's k order (B matches): lane-half hi holds contiguous K run 16*hi..16*hi+15
            a[ii].q[0] = p[2 * hi];
            a[ii].q[1] = p[2 * hi + 1];
        }
#pragma unroll
        for (int j = 0; j < 4; ++j) {
            int col = waveN * 64 + j * 16 + lo;
            // B 32x16 bf16 frag, this code's k order (same as A): lane-half hi holds contiguous K run 16*hi..16*hi+15
            const uint4* p = (const uint4*)(&lB[cur][0] + col * 32 + 16 * hi);
            FragU b;
            b.q[0] = p[0];
            b.q[1] = p[1];
#pragma unroll
            for (int ii = 0; ii < 2; ++ii) {
                acc[ii][j] = __builtin_amdgcn_wmma_f32_16x16x32_bf16(
                    false, a[ii].v, false, b.v, (short)0, acc[ii][j], false, false);
            }
        }
#if USE_TDM
        __syncthreads();           // all waves done reading before next DMA overwrites
#endif
    }

    // Epilogue. C/D 16x16 f32: VGPR r -> M = r + 8*hi, N = lane&15.
#pragma unroll
    for (int i = 0; i < 2; ++i) {
#pragma unroll
        for (int j = 0; j < 4; ++j) {
            int n = blockN + waveN * 64 + j * 16 + lo;
            float bn = bias[n];
#pragma unroll
            for (int r = 0; r < 8; ++r) {
                int m = blockM + waveM * 32 + i * 16 + r + 8 * hi;
                size_t off = (size_t)m * Ndim + n;
                float v = acc[i][j][r] + bn;
                if (RESID) outf[off] = v + resid[off];
                else       outb[off] = f2bf(v);
            }
        }
    }
}

// ---------------------------------------------------------------- attention (per-position heads x heads)
__global__ __launch_bounds__(256) void attn_kernel(const u16* __restrict__ qkv,
                                                   u16* __restrict__ ctx)
{
    int gid = blockIdx.x * blockDim.x + threadIdx.x;
    if (gid >= MROWS * NHEADS) return;
    int pos = gid / NHEADS;
    int n   = gid - pos * NHEADS;
    const u16* base = qkv + (size_t)pos * QKVN;
    const u16* q = base + n * HDIM;
    const u16* k = base + HH;
    const u16* v = base + 2 * HH;

    float qr[HDIM];
#pragma unroll
    for (int d = 0; d < HDIM; ++d) qr[d] = bf2f(q[d]);

    float sc[NHEADS];
    float mx = -1e30f;
#pragma unroll
    for (int m = 0; m < NHEADS; ++m) {
        const u16* kr = k + m * HDIM;
        float s = 0.f;
#pragma unroll
        for (int d = 0; d < HDIM; ++d) s += qr[d] * bf2f(kr[d]);
        s *= 0.125f;   // 1/sqrt(64)
        sc[m] = s;
        mx = fmaxf(mx, s);
    }
    float sum = 0.f;
#pragma unroll
    for (int m = 0; m < NHEADS; ++m) { sc[m] = __expf(sc[m] - mx); sum += sc[m]; }
    float inv = 1.f / sum;

    u16* o = ctx + (size_t)pos * HH + n * HDIM;
#pragma unroll 8
    for (int d = 0; d < HDIM; ++d) {
        float acc = 0.f;
#pragma unroll
        for (int m = 0; m < NHEADS; ++m) acc += sc[m] * bf2f(v[m * HDIM + d]);
        o[d] = f2bf(acc * inv);
    }
}

// ---------------------------------------------------------------- grouped conv as 7-tap WMMA GEMM
#define CONV_ROWS (128 + KCONV - 1)   // 134 (s-tile + halo)

__global__ __launch_bounds__(256) void conv_wmma_kernel(
    const u16* __restrict__ ctx,     // [B][S][H] bf16
    const u16* __restrict__ cw,      // [7][768][64] bf16
    const float* __restrict__ conv_b,
    u16* __restrict__ mf)            // [B*S][H] bf16
{
    __shared__ u16 lA[CONV_ROWS * HDIM];
#if USE_TDM
    __shared__ u16 lW[2][HDIM * HDIM];   // ping-pong tap weights (64x64 bf16)
#endif

    const int tid = threadIdx.x;
    const int s0  = blockIdx.x * 128;
    const int g   = blockIdx.y;
    const int b   = blockIdx.z;

#if USE_TDM
    if (tid < 32) {  // kick tap 0 weights while activations stage
        tdm_load_2d((u32)(size_t)&lW[0][0], cw + (size_t)(g * HDIM) * HDIM, 64, 64, 64);
    }
#endif

    // Stage halo'd ctx tile: rows s0-3 .. s0+130, 64 channels of group g (zero pad OOB).
    for (int c = tid; c < CONV_ROWS * 8; c += 256) {
        int row = c >> 3, seg = (c & 7) * 8;
        int s = s0 - 3 + row;
        uint4* dst = (uint4*)(lA + row * HDIM + seg);
        if (s >= 0 && s < SS) {
            const uint4* src = (const uint4*)(ctx + ((size_t)(b * SS + s) * HH + g * HDIM + seg));
            *dst = *src;
        } else {
            *dst = make_uint4(0u, 0u, 0u, 0u);
        }
    }
    __syncthreads();

    const int lane = tid & 31, wid = tid >> 5;
    const int lo = lane & 15, hi = lane >> 4;

    v8f zero = {0.f, 0.f, 0.f, 0.f, 0.f, 0.f, 0.f, 0.f};
    v8f acc[4];
#pragma unroll
    for (int j = 0; j < 4; ++j) acc[j] = zero;

    for (int t = 0; t < KCONV; ++t) {
#if USE_TDM
        if (tid < 32) {
            if (t + 1 < KCONV) {
                tdm_load_2d((u32)(size_t)&lW[(t + 1) & 1][0],
                            cw + ((size_t)(t + 1) * HH + g * HDIM) * HDIM, 64, 64, 64);
                __builtin_amdgcn_s_wait_tensorcnt(1);
            } else {
                __builtin_amdgcn_s_wait_tensorcnt(0);
            }
        }
        __syncthreads();
#endif
#pragma unroll
        for (int kk = 0; kk < 2; ++kk) {
            FragU a;
            int row = wid * 16 + lo + t;   // out row (wid*16+lo) shifted by tap
            const uint4* p = (const uint4*)(lA + row * HDIM + kk * 32);
            a.q[0] = p[2 * hi];
            a.q[1] = p[2 * hi + 1];
#pragma unroll
            for (int j = 0; j < 4; ++j) {
                FragU bf;
#if USE_TDM
                const uint4* bp =
                    (const uint4*)(&lW[t & 1][0] + ((j * 16 + lo) * HDIM + kk * 32 + 16 * hi));
#else
                int o = g * HDIM + j * 16 + lo;
                const uint4* bp =
                    (const uint4*)(cw + ((size_t)(t * HH + o) * HDIM + kk * 32 + 16 * hi));
#endif
                bf.q[0] = bp[0];
                bf.q[1] = bp[1];
                acc[j] = __builtin_amdgcn_wmma_f32_16x16x32_bf16(
                    false, a.v, false, bf.v, (short)0, acc[j], false, false);
            }
        }
#if USE_TDM
        __syncthreads();
#endif
    }

#pragma unroll
    for (int j = 0; j < 4; ++j) {
        int o = g * HDIM + j * 16 + lo;
        float cb = conv_b[o];
#pragma unroll
        for (int r = 0; r < 8; ++r) {
            int s = s0 + wid * 16 + r + 8 * hi;
            mf[(size_t)(b * SS + s) * HH + o] = f2bf(acc[j][r] + cb);
        }
    }
}

// ---------------------------------------------------------------- LayerNorm
__global__ __launch_bounds__(256) void ln_kernel(const float* __restrict__ res,
                                                 const float* __restrict__ gamma,
                                                 const float* __restrict__ beta,
                                                 float* __restrict__ out)
{
    __shared__ float red[256];
    int row = blockIdx.x;
    int t = threadIdx.x;
    const float* x = res + (size_t)row * HH;
    float v0 = x[t], v1 = x[t + 256], v2 = x[t + 512];
    red[t] = v0 + v1 + v2;
    __syncthreads();
    for (int st = 128; st > 0; st >>= 1) {
        if (t < st) red[t] += red[t + st];
        __syncthreads();
    }
    float mu = red[0] * (1.f / HH);
    __syncthreads();
    float d0 = v0 - mu, d1 = v1 - mu, d2 = v2 - mu;
    red[t] = d0 * d0 + d1 * d1 + d2 * d2;
    __syncthreads();
    for (int st = 128; st > 0; st >>= 1) {
        if (t < st) red[t] += red[t + st];
        __syncthreads();
    }
    float inv = rsqrtf(red[0] * (1.f / HH) + LN_EPS);
    float* o = out + (size_t)row * HH;
    o[t]       = gamma[t]       * d0 * inv + beta[t];
    o[t + 256] = gamma[t + 256] * d1 * inv + beta[t + 256];
    o[t + 512] = gamma[t + 512] * d2 * inv + beta[t + 512];
}

// ---------------------------------------------------------------- launch
extern "C" void kernel_launch(void* const* d_in, const int* in_sizes, int n_in,
                              void* d_out, int out_size, void* d_ws, size_t ws_size,
                              hipStream_t stream)
{
    const float* hidden = (const float*)d_in[0];
    const float* Wq     = (const float*)d_in[1];
    const float* bq     = (const float*)d_in[2];
    const float* Wk     = (const float*)d_in[3];
    const float* bk     = (const float*)d_in[4];
    const float* Wv     = (const float*)d_in[5];
    const float* bv     = (const float*)d_in[6];
    const float* conv_w = (const float*)d_in[7];
    const float* conv_b = (const float*)d_in[8];
    const float* Wo     = (const float*)d_in[9];
    const float* bo     = (const float*)d_in[10];
    const float* gamma  = (const float*)d_in[11];
    const float* beta   = (const float*)d_in[12];
    float* out = (float*)d_out;

    char* ws = (char*)d_ws;
    size_t off = 0;
    auto alloc = [&](size_t bytes) -> void* {
        void* p = ws + off;
        off = (off + bytes + 255) & ~(size_t)255;
        return p;
    };
    u16*   Xbf   = (u16*)alloc((size_t)MROWS * HH * 2);    // reused as mf after QKV GEMM
    u16*   WqkvT = (u16*)alloc((size_t)QKVN * HH * 2);
    u16*   WoT   = (u16*)alloc((size_t)HH * HH * 2);
    u16*   cw    = (u16*)alloc((size_t)KCONV * HH * HDIM * 2);
    float* bqkv  = (float*)alloc((size_t)QKVN * 4);
    u16*   qkv   = (u16*)alloc((size_t)MROWS * QKVN * 2);  // reused as res (f32, smaller) after attn
    u16*   ctx   = (u16*)alloc((size_t)MROWS * HH * 2);
    u16*   mf    = Xbf;           // X dead after QKV GEMM
    float* res   = (float*)qkv;   // qkv dead after attention; 100MB <= 151MB

    pack_x_kernel<<<(MROWS * HH + 255) / 256, 256, 0, stream>>>(hidden, Xbf, MROWS * HH);
    pack_wqkv_kernel<<<(QKVN * HH + 255) / 256, 256, 0, stream>>>(Wq, Wk, Wv, bq, bk, bv, WqkvT, bqkv);
    pack_wo_kernel<<<(HH * HH + 255) / 256, 256, 0, stream>>>(Wo, WoT);
    pack_conv_kernel<<<(HH * HDIM * KCONV + 255) / 256, 256, 0, stream>>>(conv_w, cw);

    gemm_wmma_kernel<false><<<dim3(QKVN / 128, MROWS / 128), 256, 0, stream>>>(
        Xbf, WqkvT, bqkv, nullptr, qkv, nullptr, HH, QKVN);

    attn_kernel<<<(MROWS * NHEADS + 255) / 256, 256, 0, stream>>>(qkv, ctx);

    conv_wmma_kernel<<<dim3(SS / 128, NHEADS, BB), 256, 0, stream>>>(ctx, cw, conv_b, mf);

    gemm_wmma_kernel<true><<<dim3(HH / 128, MROWS / 128), 256, 0, stream>>>(
        mf, WoT, bo, hidden, nullptr, res, HH, HH);

    ln_kernel<<<MROWS, 256, 0, stream>>>(res, gamma, beta, out);
}
